// GroupQueryAttention_16518444220876
// MI455X (gfx1250) — compile-verified, matrix operand order checked by reading, not run on hardware
//
#include <hip/hip_runtime.h>
#include <hip/hip_bf16.h>
#include <stdint.h>

#define B_SZ     2
#define S_LEN    2048
#define D_MODEL  2048
#define N_HEADS  32
#define N_GROUPS 8
#define HEAD_DIM 64
#define KV_DIM   (N_GROUPS * HEAD_DIM)   // 512
#define M_ROWS   (B_SZ * S_LEN)          // 4096

typedef __attribute__((ext_vector_type(16))) __bf16 v16bf;
typedef __attribute__((ext_vector_type(8)))  __bf16 v8bf;
typedef __attribute__((ext_vector_type(8)))  float  v8f;

static __device__ __forceinline__ unsigned short f2bf(float f) {
  union { float f; unsigned u; } c; c.f = f;
  unsigned u = c.u + 0x7FFFu + ((c.u >> 16) & 1u);   // round-to-nearest-even
  return (unsigned short)(u >> 16);
}

static __device__ __forceinline__ v16bf cat16(v8bf lo, v8bf hi) {
  return __builtin_shufflevector(lo, hi, 0,1,2,3,4,5,6,7,8,9,10,11,12,13,14,15);
}

// Row-contiguous bf16 fragment load implementing the CDNA5 16-bit A/B VGPR
// layout: lanes 0-15 hold K=[0..7]&[16..23], lanes 16-31 hold K=[8..15]&[24..31].
static __device__ __forceinline__ v16bf load_frag(const unsigned short* p, int hsel) {
  const v8bf lo = *reinterpret_cast<const v8bf*>(p + hsel * 8);
  const v8bf hi = *reinterpret_cast<const v8bf*>(p + hsel * 8 + 16);
  return cat16(lo, hi);
}

static __device__ __forceinline__ v8f wmma_bf16(v16bf a, v16bf b, v8f c) {
  return __builtin_amdgcn_wmma_f32_16x16x32_bf16(false, a, false, b,
                                                 (short)0, c, false, false);
}

// ---------------------------------------------------------------- converts
__global__ __launch_bounds__(256) void cvt_bf16_kernel(
    const float* __restrict__ in, unsigned short* __restrict__ out, int n) {
  for (int i = blockIdx.x * blockDim.x + threadIdx.x; i < n;
       i += gridDim.x * blockDim.x)
    out[i] = f2bf(in[i]);
}

// W[K,N] fp32 -> Wt[N,K] bf16 (coalesced via LDS tile)
__global__ __launch_bounds__(256) void transpose_bf16_kernel(
    const float* __restrict__ W, unsigned short* __restrict__ Wt, int K, int N) {
  __shared__ float tile[32][33];
  const int k0 = blockIdx.x * 32, n0 = blockIdx.y * 32;
  const int tx = threadIdx.x & 31, ty = threadIdx.x >> 5;
  #pragma unroll
  for (int i = ty; i < 32; i += 8)
    tile[i][tx] = W[(size_t)(k0 + i) * N + n0 + tx];
  __syncthreads();
  #pragma unroll
  for (int i = ty; i < 32; i += 8)
    Wt[(size_t)(n0 + i) * K + k0 + tx] = f2bf(tile[tx][i]);
}

// ---------------------------------------------------------------- GEMM
// C[M,N] = A[M,K] @ W  (W given transposed: Wt[N,K]) + bias
// Per wave: 16 x 128 C tile (8 N-tiles) -> 8 WMMAs per A-fragment load.
// MODE 0: bf16 row-major out; MODE 1: bf16 V-transposed out (Vt[b,g,d,s]);
// MODE 2: fp32 row-major out.
template <int MODE>
__global__ __launch_bounds__(256) void gemm_bf16_kernel(
    const unsigned short* __restrict__ A,
    const unsigned short* __restrict__ Wt,
    const float* __restrict__ bias,
    void* __restrict__ out, int M, int N, int K) {
  const int lane = threadIdx.x & 31;
  const int wave = threadIdx.x >> 5;       // 8 waves, M-split (shared B frags)
  const int lr   = lane & 15;
  const int hsel = lane >> 4;
  const int m0 = blockIdx.x * 128 + wave * 16;
  const int n0 = blockIdx.y * 128;

  const unsigned short* Arow = A + (size_t)(m0 + lr) * K;
  v8f acc[8] = {};

  for (int k0 = 0; k0 < K; k0 += 32) {
    __builtin_prefetch(Arow + k0 + 128, 0, 0);       // global_prefetch_b8
    const v16bf af = load_frag(Arow + k0, hsel);
    #pragma unroll
    for (int t = 0; t < 8; ++t) {
      const v16bf bf = load_frag(Wt + (size_t)(n0 + t * 16 + lr) * K + k0, hsel);
      acc[t] = wmma_bf16(af, bf, acc[t]);
    }
  }

  #pragma unroll
  for (int t = 0; t < 8; ++t) {
    const int col = n0 + t * 16 + lr;
    const float bv = bias[col];
    #pragma unroll
    for (int v = 0; v < 8; ++v) {
      const int row = m0 + v + 8 * hsel;      // C layout: M = vgpr + 8*half
      const float val = acc[t][v] + bv;
      if (MODE == 0) {
        ((unsigned short*)out)[(size_t)row * N + col] = f2bf(val);
      } else if (MODE == 2) {
        ((float*)out)[(size_t)row * N + col] = val;
      } else {  // V transposed: Vt[((b*G+g)*64+d)*S + s]
        const int b = row / S_LEN, s = row % S_LEN;
        const int g = col >> 6, d = col & 63;
        ((unsigned short*)out)[((size_t)(b * N_GROUPS + g) * HEAD_DIM + d) * S_LEN + s]
            = f2bf(val);
      }
    }
  }
}

// ---------------------------------------------------------------- attention
// One wave = 16 query rows of one head; flash loop over keys in 64-blocks:
// 16 WMMAs per iteration (8 score + 8 ctx), shuffle-reductions amortized 4-wide.
__global__ __launch_bounds__(256) void attn_kernel(
    const unsigned short* __restrict__ Q,     // [B,S,D] bf16
    const unsigned short* __restrict__ Km,    // [B,S,512] bf16
    const unsigned short* __restrict__ Vt,    // [B,G,64,S] bf16
    const int* __restrict__ mask,             // [B,S]
    unsigned short* __restrict__ ctx) {       // [B,S,D] bf16
  __shared__ unsigned short Pbuf[8][16][72];  // pitch 72: conflict-free 16B reads

  const int lane = threadIdx.x & 31;
  const int wave = threadIdx.x >> 5;
  const int lr   = lane & 15;
  const int hsel = lane >> 4;

  int flat = blockIdx.x * 8 + wave;                  // B*H*(S/16) waves total
  const int qb = flat % (S_LEN / 16); flat /= (S_LEN / 16);
  const int h  = flat % N_HEADS;
  const int b  = flat / N_HEADS;
  const int g  = h >> 2;                             // HPG = 4

  // Q fragments (constant across the key loop)
  const unsigned short* Qrow =
      Q + ((size_t)(b * S_LEN + qb * 16 + lr)) * D_MODEL + h * HEAD_DIM;
  const v16bf qf0 = load_frag(Qrow, hsel);
  const v16bf qf1 = load_frag(Qrow + 32, hsel);

  const unsigned short* Kbase = Km + (size_t)b * S_LEN * KV_DIM + g * HEAD_DIM;
  const unsigned short* Vbase =
      Vt + (size_t)(b * N_GROUPS + g) * HEAD_DIM * S_LEN;
  const int* mrow = mask + b * S_LEN;

  float m[8], ssum[8];
  v8f acc[4] = {};
  #pragma unroll
  for (int v = 0; v < 8; ++v) { m[v] = -__builtin_inff(); ssum[v] = 0.f; }
  const float scale = 0.125f;                        // 1/sqrt(64)

  for (int kb = 0; kb < S_LEN; kb += 64) {
    // ---- scores: four 16x16 key tiles, K=64 split into 2 wmma steps each
    v8f s[4];
    float mk[4];
    #pragma unroll
    for (int t = 0; t < 4; ++t) {
      const unsigned short* Kr = Kbase + (size_t)(kb + t * 16 + lr) * KV_DIM;
      __builtin_prefetch(Kr + 256 * KV_DIM, 0, 0);
      v8f st = {};
      st = wmma_bf16(qf0, load_frag(Kr, hsel), st);
      st = wmma_bf16(qf1, load_frag(Kr + 32, hsel), st);
      s[t] = st;
      mk[t] = (mrow[kb + t * 16 + lr] == 0) ? -__builtin_inff() : 0.f;
    }

    // ---- online softmax in C-register layout (row = v + 8*hsel)
    #pragma unroll
    for (int v = 0; v < 8; ++v) {
      float a0 = s[0][v] * scale + mk[0];
      float a1 = s[1][v] * scale + mk[1];
      float a2 = s[2][v] * scale + mk[2];
      float a3 = s[3][v] * scale + mk[3];
      float rm = fmaxf(fmaxf(a0, a1), fmaxf(a2, a3));
      rm = fmaxf(rm, __shfl_xor(rm, 1, 32));
      rm = fmaxf(rm, __shfl_xor(rm, 2, 32));
      rm = fmaxf(rm, __shfl_xor(rm, 4, 32));
      rm = fmaxf(rm, __shfl_xor(rm, 8, 32));
      const float nm = fmaxf(m[v], rm);
      float alpha, p0, p1, p2, p3;
      if (nm == -__builtin_inff()) {
        alpha = 1.f; p0 = p1 = p2 = p3 = 0.f;
      } else {
        alpha = __expf(m[v] - nm);
        p0 = __expf(a0 - nm);
        p1 = __expf(a1 - nm);
        p2 = __expf(a2 - nm);
        p3 = __expf(a3 - nm);
      }
      m[v] = nm;
      float rs = (p0 + p1) + (p2 + p3);
      rs += __shfl_xor(rs, 1, 32);
      rs += __shfl_xor(rs, 2, 32);
      rs += __shfl_xor(rs, 4, 32);
      rs += __shfl_xor(rs, 8, 32);
      ssum[v] = ssum[v] * alpha + rs;
      #pragma unroll
      for (int t = 0; t < 4; ++t) acc[t][v] *= alpha;
      s[0][v] = p0; s[1][v] = p1; s[2][v] = p2; s[3][v] = p3;
    }

    // ---- re-layout P (C-layout) -> A-fragments via per-wave LDS tile
    #pragma unroll
    for (int v = 0; v < 8; ++v) {
      const int r = v + 8 * hsel;
      #pragma unroll
      for (int t = 0; t < 4; ++t)
        Pbuf[wave][r][t * 16 + lr] = f2bf(s[t][v]);
    }
    asm volatile("s_wait_dscnt 0x0" ::: "memory");   // CDNA5 split DS counter
    const unsigned short* prow = &Pbuf[wave][lr][0];

    // ---- ctx += P @ V  (two 32-key chunks; Vt rows contiguous along keys)
    #pragma unroll
    for (int c = 0; c < 2; ++c) {
      const v16bf pf =
          cat16(*reinterpret_cast<const v8bf*>(prow + c * 32 + hsel * 8),
                *reinterpret_cast<const v8bf*>(prow + c * 32 + hsel * 8 + 16));
      #pragma unroll
      for (int t = 0; t < 4; ++t) {
        const unsigned short* Vr =
            Vbase + (size_t)(t * 16 + lr) * S_LEN + kb + c * 32;
        acc[t] = wmma_bf16(pf, load_frag(Vr, hsel), acc[t]);
      }
    }
  }

  // ---- normalize (reciprocal-multiply) + store bf16 ctx
  #pragma unroll
  for (int v = 0; v < 8; ++v) {
    const float d   = ssum[v];
    const float inv = (d > 0.f) ? 1.f / d : 0.f;
    const int row = qb * 16 + v + 8 * hsel;
    #pragma unroll
    for (int t = 0; t < 4; ++t) {
      const int col = h * HEAD_DIM + t * 16 + lr;
      ctx[((size_t)(b * S_LEN) + row) * D_MODEL + col] = f2bf(acc[t][v] * inv);
    }
  }
}

// ---------------------------------------------------------------- launch
extern "C" void kernel_launch(void* const* d_in, const int* in_sizes, int n_in,
                              void* d_out, int out_size, void* d_ws, size_t ws_size,
                              hipStream_t stream) {
  (void)in_sizes; (void)n_in; (void)out_size; (void)ws_size;
  const float* x    = (const float*)d_in[0];
  const int*   mask = (const int*)  d_in[1];
  const float* q_w  = (const float*)d_in[2];
  const float* q_b  = (const float*)d_in[3];
  const float* k_w  = (const float*)d_in[4];
  const float* k_b  = (const float*)d_in[5];
  const float* v_w  = (const float*)d_in[6];
  const float* v_b  = (const float*)d_in[7];
  const float* o_w  = (const float*)d_in[8];
  const float* o_b  = (const float*)d_in[9];
  float* out = (float*)d_out;

  char* w = (char*)d_ws;
  unsigned short* xbf = (unsigned short*)w; w += (size_t)M_ROWS * D_MODEL * 2;
  unsigned short* qwt = (unsigned short*)w; w += (size_t)D_MODEL * D_MODEL * 2;
  unsigned short* kwt = (unsigned short*)w; w += (size_t)KV_DIM  * D_MODEL * 2;
  unsigned short* vwt = (unsigned short*)w; w += (size_t)KV_DIM  * D_MODEL * 2;
  unsigned short* owt = (unsigned short*)w; w += (size_t)D_MODEL * D_MODEL * 2;
  unsigned short* Qb  = (unsigned short*)w; w += (size_t)M_ROWS * D_MODEL * 2;
  unsigned short* Kb  = (unsigned short*)w; w += (size_t)M_ROWS * KV_DIM  * 2;
  unsigned short* Vtb = (unsigned short*)w; w += (size_t)M_ROWS * KV_DIM  * 2;
  unsigned short* ctx = (unsigned short*)w; w += (size_t)M_ROWS * D_MODEL * 2;

  // 1) precision conversion / weight transposes
  cvt_bf16_kernel<<<4096, 256, 0, stream>>>(x, xbf, M_ROWS * D_MODEL);
  transpose_bf16_kernel<<<dim3(64, 64), 256, 0, stream>>>(q_w, qwt, D_MODEL, D_MODEL);
  transpose_bf16_kernel<<<dim3(64, 16), 256, 0, stream>>>(k_w, kwt, D_MODEL, KV_DIM);
  transpose_bf16_kernel<<<dim3(64, 16), 256, 0, stream>>>(v_w, vwt, D_MODEL, KV_DIM);
  transpose_bf16_kernel<<<dim3(64, 64), 256, 0, stream>>>(o_w, owt, D_MODEL, D_MODEL);

  // 2) projections (bf16 WMMA, fp32 accumulate)
  gemm_bf16_kernel<0><<<dim3(32, 16), 256, 0, stream>>>(xbf, qwt, q_b, Qb,
                                                        M_ROWS, D_MODEL, D_MODEL);
  gemm_bf16_kernel<0><<<dim3(32, 4), 256, 0, stream>>>(xbf, kwt, k_b, Kb,
                                                       M_ROWS, KV_DIM, D_MODEL);
  gemm_bf16_kernel<1><<<dim3(32, 4), 256, 0, stream>>>(xbf, vwt, v_b, Vtb,
                                                       M_ROWS, KV_DIM, D_MODEL);

  // 3) fused flash attention (B*H*S/16 = 8192 waves)
  attn_kernel<<<1024, 256, 0, stream>>>(Qb, Kb, Vtb, mask, ctx);

  // 4) output projection, fp32 out
  gemm_bf16_kernel<2><<<dim3(32, 16), 256, 0, stream>>>(ctx, owt, o_b, out,
                                                        M_ROWS, D_MODEL, D_MODEL);
}
